// DehazeNet_30434138260002
// MI455X (gfx1250) — matrix operand order repaired and checked by reading, not compiled, not run on hardware
//
#include <hip/hip_runtime.h>
#include <hip/hip_bf16.h>

// ---------------------------------------------------------------------------
// DehazeNet on MI455X (gfx1250, wave32, WMMA).
//
// Pipeline (all on `stream`, deterministic):
//   1) batch_max_kernel        : x_g[b] = max(x[b])   (softmax over size-1 axis == 1)
//   2) per stage s in {p=1,p=2,p=3}:
//        pack_b_kernel         : w_qkv (F,L) f32 -> f16 packed in WMMA B lane layout
//        attn_stage_kernel<P,H>: gather(reflect) -> v_wmma_f32_16x16x32_f16 GEMM
//                                -> qkv in LDS(f16) -> softmax attention
//                                -> head mix (ds_add_f32) -> scatter to image
//   3) conv_final_kernel       : 5x5 conv over concat(x9,x6,x3), relu,
//                                out = relu(x*x0 + (x_g - x0))
// ---------------------------------------------------------------------------

typedef __attribute__((ext_vector_type(16))) _Float16 v16h;
typedef __attribute__((ext_vector_type(8)))  float    v8f;

#define BATCH 4
#define CH    3
#define IMH   256
#define IMW   256

// ---------------------------------------------------------------------------
// x_g[b] = max over (c,h,w) of x[b]
// ---------------------------------------------------------------------------
__global__ __launch_bounds__(256) void batch_max_kernel(
    const float* __restrict__ x, float* __restrict__ xg)
{
    __shared__ float red[256];
    const int b = blockIdx.x;
    const float* p = x + (size_t)b * CH * IMH * IMW;
    float m = -1e30f;
    for (int i = threadIdx.x; i < CH * IMH * IMW; i += 256)
        m = fmaxf(m, p[i]);
    red[threadIdx.x] = m;
    __syncthreads();
    for (int s = 128; s > 0; s >>= 1) {
        if (threadIdx.x < s)
            red[threadIdx.x] = fmaxf(red[threadIdx.x], red[threadIdx.x + s]);
        __syncthreads();
    }
    if (threadIdx.x == 0) xg[b] = red[0];
}

// ---------------------------------------------------------------------------
// Pack w_qkv (row-major (F,L), f32) into f16 WMMA-B layout:
//   Bp[(ft*32 + lane)*16 + i] = W[f][k],  f = ft*16 + (lane&15),
//                                         k = (lane>>4)*16 + i,  0 if OOB.
// ---------------------------------------------------------------------------
__global__ void pack_b_kernel(const float* __restrict__ w,
                              _Float16* __restrict__ Bp,
                              int F, int L, int Fc)
{
    int idx = blockIdx.x * blockDim.x + threadIdx.x;
    int total = Fc * 512;
    if (idx >= total) return;
    int ft   = idx >> 9;
    int r    = idx & 511;
    int lane = r >> 4;
    int i    = r & 15;
    int f = ft * 16 + (lane & 15);
    int k = (lane >> 4) * 16 + i;
    float v = (f < F && k < L) ? w[f * L + k] : 0.0f;
    Bp[idx] = (_Float16)v;
}

// ---------------------------------------------------------------------------
// Fused windowed-attention stage.
//   Block = 288 threads = 9 waves; covers 16 window positions (n).
//   GEMM rows: 144 = 16 n * 9 neighbors; wave w owns the 16-row tile w.
//   K = L = 3*P*P (<= 27), zero-padded to 32 -> one WMMA per 16x16 out tile.
// ---------------------------------------------------------------------------
template <int P, int HEADS>
__global__ __launch_bounds__(288) void attn_stage_kernel(
    const float*    __restrict__ in,      // (4,3,256,256)
    const _Float16* __restrict__ Bpack,   // Fc*512 packed weights
    const float*    __restrict__ b_qkv,   // (F,)
    const float*    __restrict__ head_w,  // (1,HEADS) -> head_w[hd]
    float*          __restrict__ out)     // (4,3,256,256)
{
    constexpr int PP   = P * P;
    constexpr int L    = CH * PP;                        // 3, 12, 27
    constexpr int F    = HEADS * 3 * CH * PP;            // 54, 144, 162
    constexpr int Fc   = (F + 15) / 16;                  // 4, 9, 11
    constexpr int Fpad = Fc * 16;
    constexpr int PAD  = (P * (1 + IMH / P) - IMH) % P;  // 0, 0, 2
    constexpr int NH   = (IMH + PAD) / P;                // 256, 128, 86
    constexpr int NW   = (IMW + PAD) / P;

    __shared__ _Float16 qkv_lds[144 * Fpad];  // stage3: 50688 B
    __shared__ float    outacc[16 * L];

    const int t    = threadIdx.x;
    const int lane = t & 31;
    const int wave = t >> 5;       // 0..8 -> row tile
    const int hi   = lane >> 4;    // lane half
    const int col  = lane & 15;

    const int nbase = blockIdx.x * 16;

    for (int i = t; i < 16 * L; i += 288) outacc[i] = 0.0f;

    // ---------------- GEMM phase: qkv = A @ W^T + b ----------------
    // This lane's A row (rows are (n, jwin) pairs).
    const int row_local = wave * 16 + col;
    const int n_local   = row_local / 9;
    const int jwin      = row_local % 9;
    const int n  = nbase + n_local;
    const int bb = n / (NH * NW);
    const int r2 = n % (NH * NW);
    const int ih = r2 / NW;
    const int iw = r2 % NW;
    const int di = jwin / 3, dj = jwin % 3;

    __builtin_prefetch(&in[((size_t)(bb * CH) * IMH + (size_t)(ih * P)) * IMW + iw * P], 0, 3);

    v16h a;
#pragma unroll
    for (int i = 0; i < 16; ++i) {
        const int k = hi * 16 + i;
        float v = 0.0f;
        if (k < L) {
            const int cc = k / PP, rm = k % PP;
            const int py = rm / P, px = rm % P;
            int y = (ih + di) * P + py - (P + PAD);
            int x = (iw + dj) * P + px - (P + PAD);
            y = (y < 0) ? -y : ((y >= IMH) ? 2 * IMH - 2 - y : y);
            x = (x < 0) ? -x : ((x >= IMW) ? 2 * IMW - 2 - x : x);
            v = in[((size_t)(bb * CH + cc) * IMH + y) * IMW + x];
        }
        a[i] = (_Float16)v;
    }

    for (int ft = 0; ft < Fc; ++ft) {
        v16h bv = *(const v16h*)(Bpack + (ft * 32 + lane) * 16);
        v8f  c  = {};
        c = __builtin_amdgcn_wmma_f32_16x16x32_f16(
                /*neg_a=*/false, a, /*neg_b=*/false, bv,
                /*c_mod=*/(short)0, c, /*reuse_a=*/false, /*reuse_b=*/false);
        const int f   = ft * 16 + col;
        const float bq = (f < F) ? b_qkv[f] : 0.0f;
#pragma unroll
        for (int vv = 0; vv < 8; ++vv) {
            const int rl = wave * 16 + vv + 8 * hi;   // C row = vv + 8*hi
            qkv_lds[rl * Fpad + f] = (_Float16)(c[vv] + bq);
        }
    }
    __syncthreads();

    // ---------------- Attention phase ----------------
    // feature index: f = ((hd*3 + which)*CH + cc)*PP + pi   (which: 0=q,1=k,2=v)
    constexpr int TASKS = 16 * HEADS * CH;
    if (t < TASKS) {
        const int nl = t / (HEADS * CH);
        const int rr = t % (HEADS * CH);
        const int hd = rr / CH;
        const int cc = rr % CH;
        const float inv_sqrtL = rsqrtf((float)L);

        float q4[PP];
#pragma unroll
        for (int pi = 0; pi < PP; ++pi)
            q4[pi] = (float)qkv_lds[(nl * 9 + 4) * Fpad + ((hd * 3 + 0) * CH + cc) * PP + pi];

        float sc[9];
        float m = -1e30f;
        for (int j = 0; j < 9; ++j) {
            const int dii = j / 3, dji = j % 3;
            float s = 0.0f;
#pragma unroll
            for (int pi = 0; pi < PP; ++pi) {
                const int py = pi / P, px = pi % P;
                const int ay = (dii == 0) ? (P - 1 - py) : ((dii == 1) ? 0 : py);
                const int ax = (dji == 0) ? (P - 1 - px) : ((dji == 1) ? 0 : px);
                const float bia = __expf(-(float)(ay + ax) / (float)P);
                const float kv =
                    (float)qkv_lds[(nl * 9 + j) * Fpad + ((hd * 3 + 1) * CH + cc) * PP + pi];
                s += q4[pi] * kv * bia;
            }
            sc[j] = s * inv_sqrtL;
            m = fmaxf(m, sc[j]);
        }
        float den = 0.0f;
        for (int j = 0; j < 9; ++j) { sc[j] = __expf(sc[j] - m); den += sc[j]; }
        const float hw = head_w[hd] / den;   // head mix weight folded with softmax norm

        float o[PP];
#pragma unroll
        for (int pi = 0; pi < PP; ++pi) o[pi] = 0.0f;
        for (int j = 0; j < 9; ++j) {
#pragma unroll
            for (int pi = 0; pi < PP; ++pi) {
                const float vv =
                    (float)qkv_lds[(nl * 9 + j) * Fpad + ((hd * 3 + 2) * CH + cc) * PP + pi];
                o[pi] += sc[j] * vv;
            }
        }
#pragma unroll
        for (int pi = 0; pi < PP; ++pi)
            atomicAdd(&outacc[nl * L + cc * PP + pi], o[pi] * hw);  // ds_add_f32
    }
    __syncthreads();

    // ---------------- Scatter (with crop) ----------------
    for (int idx = t; idx < 16 * L; idx += 288) {
        const int nl = idx / L;
        const int d  = idx % L;
        const int cc = d / PP, rm = d % PP;
        const int py = rm / P, px = rm % P;
        const int nn  = nbase + nl;
        const int bb2 = nn / (NH * NW);
        const int r3  = nn % (NH * NW);
        const int ih2 = r3 / NW, iw2 = r3 % NW;
        const int Y = ih2 * P + py - PAD;
        const int X = iw2 * P + px - PAD;
        if (Y >= 0 && X >= 0)
            out[((size_t)(bb2 * CH + cc) * IMH + Y) * IMW + X] = outacc[idx];
    }
}

// ---------------------------------------------------------------------------
// x0 = relu(conv5x5(concat(x9,x6,x3)) + cb);  out = relu(x*x0 + (x_g - x0))
// grid = (H*W/256, 3, 4)
// ---------------------------------------------------------------------------
__global__ __launch_bounds__(256) void conv_final_kernel(
    const float* __restrict__ x9, const float* __restrict__ x6,
    const float* __restrict__ x3, const float* __restrict__ xin,
    const float* __restrict__ xg, const float* __restrict__ cw,
    const float* __restrict__ cb, float* __restrict__ out)
{
    const int oc = blockIdx.y;
    const int bb = blockIdx.z;
    __shared__ float wsm[9 * 25];
    for (int i = threadIdx.x; i < 225; i += 256) wsm[i] = cw[oc * 225 + i];
    __syncthreads();

    const int pix = blockIdx.x * 256 + threadIdx.x;
    const int y = pix / IMW, xcol = pix % IMW;

    float acc = cb[oc];
    for (int icg = 0; icg < 3; ++icg) {
        const float* src = (icg == 0) ? x9 : ((icg == 1) ? x6 : x3);
        src += (size_t)bb * CH * IMH * IMW;
        for (int c = 0; c < CH; ++c) {
            const int ic = icg * 3 + c;
            const float* sp = src + (size_t)c * IMH * IMW;
#pragma unroll
            for (int ky = 0; ky < 5; ++ky) {
                const int yy = y + ky - 2;
                if (yy < 0 || yy >= IMH) continue;
#pragma unroll
                for (int kx = 0; kx < 5; ++kx) {
                    const int xx = xcol + kx - 2;
                    if (xx < 0 || xx >= IMW) continue;
                    acc += wsm[ic * 25 + ky * 5 + kx] * sp[yy * IMW + xx];
                }
            }
        }
    }
    const float x0 = fmaxf(acc, 0.0f);
    const size_t o = ((size_t)(bb * CH + oc) * IMH + y) * IMW + xcol;
    const float xv = xin[o];
    out[o] = fmaxf(xv * x0 + (xg[bb] - x0), 0.0f);
}

// ---------------------------------------------------------------------------
// Launch
// ---------------------------------------------------------------------------
extern "C" void kernel_launch(void* const* d_in, const int* in_sizes, int n_in,
                              void* d_out, int out_size, void* d_ws, size_t ws_size,
                              hipStream_t stream)
{
    (void)in_sizes; (void)n_in; (void)out_size; (void)ws_size;

    const float* x   = (const float*)d_in[0];
    const float* w3  = (const float*)d_in[1];
    const float* b3  = (const float*)d_in[2];
    const float* hw3 = (const float*)d_in[3];
    const float* w6  = (const float*)d_in[4];
    const float* b6  = (const float*)d_in[5];
    const float* hw6 = (const float*)d_in[6];
    const float* w9  = (const float*)d_in[7];
    const float* b9  = (const float*)d_in[8];
    const float* hw9 = (const float*)d_in[9];
    const float* cw  = (const float*)d_in[10];
    const float* cb  = (const float*)d_in[11];

    const size_t IMG = (size_t)BATCH * CH * IMH * IMW;   // 786432 floats
    float* buf3 = (float*)d_ws;
    float* buf6 = buf3 + IMG;
    float* buf9 = buf6 + IMG;
    float* xg   = buf9 + IMG;
    _Float16* Bp1 = (_Float16*)(xg + 4);
    _Float16* Bp2 = Bp1 + 4 * 512;
    _Float16* Bp3 = Bp2 + 9 * 512;

    batch_max_kernel<<<BATCH, 256, 0, stream>>>(x, xg);

    // Stage 1: p=1, heads=6, L=3, F=54, Fc=4, N=4*256*256
    pack_b_kernel<<<(4 * 512 + 255) / 256, 256, 0, stream>>>(w3, Bp1, 54, 3, 4);
    attn_stage_kernel<1, 6><<<(4 * 256 * 256) / 16, 288, 0, stream>>>(x, Bp1, b3, hw3, buf3);

    // Stage 2: p=2, heads=4, L=12, F=144, Fc=9, N=4*128*128
    pack_b_kernel<<<(9 * 512 + 255) / 256, 256, 0, stream>>>(w6, Bp2, 144, 12, 9);
    attn_stage_kernel<2, 4><<<(4 * 128 * 128) / 16, 288, 0, stream>>>(buf3, Bp2, b6, hw6, buf6);

    // Stage 3: p=3, heads=2, L=27, F=162, Fc=11, N=4*86*86
    pack_b_kernel<<<(11 * 512 + 255) / 256, 256, 0, stream>>>(w9, Bp3, 162, 27, 11);
    attn_stage_kernel<3, 2><<<(4 * 86 * 86) / 16, 288, 0, stream>>>(buf6, Bp3, b9, hw9, buf9);

    conv_final_kernel<<<dim3((IMH * IMW) / 256, 3, BATCH), 256, 0, stream>>>(
        buf9, buf6, buf3, x, xg, cw, cb, (float*)d_out);
}
